// GatedDynamicFusion_47218870452765
// MI455X (gfx1250) — matrix operand order repaired and checked by reading, not compiled, not run on hardware
//
#include <hip/hip_runtime.h>
#include <hip/hip_bf16.h>
#include <math.h>

// ---------------------------------------------------------------------------
// GatedDynamicFusion on MI455X (gfx1250, wave32).
// Heavy GEMMs run on v_wmma_f32_16x16x32_bf16 (f32 accumulate).
// softmax(axis=0).mean(axis=0) == 1/N identity -> global_w = tanh(1/N) const,
// so the A_base @ E_cur GEMM + column softmax are eliminated entirely.
// v2: register-staged double-buffered GEMM pipeline, batched B-fragment
//     loads before the WMMA burst, boundary guard templated out of the
//     M%128==0 launches.
// ---------------------------------------------------------------------------

#define BB 64
#define TT 256
#define HH 1024
#define NN 1024

typedef __bf16 bf16_t;
typedef __attribute__((ext_vector_type(16))) __bf16 v16bf;
typedef __attribute__((ext_vector_type(8)))  __bf16 v8bf;
typedef __attribute__((ext_vector_type(8)))  float  v8f;

// ---------------------------------------------------------------------------
// Kernel 1: per (b,h): h_gate = mean_t h_time, and bf16 copy of h_time.
// Fuses the f32->bf16 conversion into the mandatory 64MB read.
// grid = (H/256, B), block = 256
// ---------------------------------------------------------------------------
__global__ __launch_bounds__(256) void reduce_bt_kernel(
    const float* __restrict__ h, float* __restrict__ hgate_f,
    bf16_t* __restrict__ hgate_b, bf16_t* __restrict__ h_b) {
  int hi = blockIdx.x * 256 + threadIdx.x;
  int b  = blockIdx.y;
  size_t base = ((size_t)b * TT) * HH + hi;
  float s = 0.0f;
  for (int t = 0; t < TT; ++t) {
    float v = h[base + (size_t)t * HH];
    s += v;
    h_b[base + (size_t)t * HH] = (bf16_t)v;
  }
  float m = s * (1.0f / TT);
  hgate_f[b * HH + hi] = m;
  hgate_b[b * HH + hi] = (bf16_t)m;
}

// h_mean = mean_b h_gate.  grid = H/256
__global__ __launch_bounds__(256) void reduce_b_kernel(
    const float* __restrict__ hgate_f, float* __restrict__ hmean) {
  int hi = blockIdx.x * 256 + threadIdx.x;
  float s = 0.0f;
  for (int b = 0; b < BB; ++b) s += hgate_f[b * HH + hi];
  hmean[hi] = s * (1.0f / BB);
}

// generic f32 -> bf16 weight conversion
__global__ void cvt_bf16_kernel(const float* __restrict__ in,
                                bf16_t* __restrict__ out, int n) {
  int i = blockIdx.x * blockDim.x + threadIdx.x;
  if (i < n) out[i] = (bf16_t)in[i];
}

// ---------------------------------------------------------------------------
// Kernel 2: E = E_dyn + h_mean; row L2-normalize; emit bf16 E_cur [N,H] and
// its transpose [H,N] so the GEMM kernel only needs row-major operands.
// grid = N, block = 256 (4 elems/thread)
// ---------------------------------------------------------------------------
__global__ __launch_bounds__(256) void ecur_kernel(
    const float* __restrict__ E_dyn, const float* __restrict__ hmean,
    bf16_t* __restrict__ Ecur, bf16_t* __restrict__ EcurT) {
  __shared__ float red[256];
  int n = blockIdx.x, tid = threadIdx.x;
  float e[4]; float ss = 0.0f;
#pragma unroll
  for (int i = 0; i < 4; ++i) {
    int hi = tid + i * 256;
    float v = E_dyn[(size_t)n * HH + hi] + hmean[hi];
    e[i] = v; ss += v * v;
  }
  red[tid] = ss; __syncthreads();
  for (int off = 128; off > 0; off >>= 1) {
    if (tid < off) red[tid] += red[tid + off];
    __syncthreads();
  }
  float inv = 1.0f / fmaxf(sqrtf(red[0]), 1e-12f);
#pragma unroll
  for (int i = 0; i < 4; ++i) {
    int hi = tid + i * 256;
    float v = e[i] * inv;
    Ecur[(size_t)n * HH + hi]  = (bf16_t)v;
    EcurT[(size_t)hi * NN + n] = (bf16_t)v;
  }
}

// ---------------------------------------------------------------------------
// WMMA bf16 GEMM: C[M,N] = act(A[M,K] @ B[K,N] + bias), f32 accumulate.
// Tile 128x128x32; 8 waves/block; wave w owns rows [16w,16w+16) x 128 N.
// Pipeline: K-tile k+1 is fetched into registers while the 8 WMMAs of
// tile k execute; LDS single-buffered with two barriers per step.
// B staged transposed (Bs[n][k]) so one B fragment = two 16-byte k runs
// per lane (2x ds_load_b128). 80-byte padded LDS rows keep 16B alignment
// and spread banks. All 8 B fragments are loaded before the WMMA burst.
// Fragment layouts follow ISA 7.12.2 (16-bit A 16x32, B 32x16, f32 C 16x16).
// ---------------------------------------------------------------------------
#define ACT_NONE 0
#define ACT_RELU 1
#define ACT_GELU 2
#define ACT_SIG  3

template <int ACT, bool GUARD>
__global__ __launch_bounds__(256) void gemm_bf16_kernel(
    const bf16_t* __restrict__ A, const bf16_t* __restrict__ Bm,
    const float* __restrict__ bias,
    float* __restrict__ Cf, bf16_t* __restrict__ Cb,
    int M, int N, int K) {
  constexpr int BM = 128, BN = 128, BK = 32, LDP = 40;  // 80B LDS row stride
  __shared__ bf16_t As[BM * LDP];   // [m][k]
  __shared__ bf16_t Bs[BN * LDP];   // [n][k] (transposed)

  const int tid = threadIdx.x;
  const int wv = tid >> 5, lane = tid & 31;
  const int m0 = blockIdx.y * BM, n0 = blockIdx.x * BN;

  // per-thread staging coordinates (32B of A, 32B of B per K-step)
  const int rowA = tid >> 1;            // 0..127
  const int colA = (tid & 1) * 16;      // 0 or 16
  const int kB   = tid >> 3;            // 0..31
  const int nB   = (tid & 7) * 16;      // 0..112

  v8f acc[8];
#pragma unroll
  for (int i = 0; i < 8; ++i)
#pragma unroll
    for (int j = 0; j < 8; ++j) acc[i][j] = 0.0f;

  const int lane15 = lane & 15;
  const int kbA = (lane < 16) ? 0 : 8;    // A-frag K base (ISA layout)
  const int kbB = (lane < 16) ? 0 : 8;    // B-frag K base
  const int mrow = wv * 16 + lane15;      // local A row this lane owns

  auto loadA = [&](int k0, v8bf& x0, v8bf& x1) {
    if (!GUARD || (m0 + rowA) < M) {
      const bf16_t* p = A + (size_t)(m0 + rowA) * K + k0 + colA;
      x0 = *(const v8bf*)(p);
      x1 = *(const v8bf*)(p + 8);
    } else {
#pragma unroll
      for (int e = 0; e < 8; ++e) { x0[e] = (bf16_t)0.0f; x1[e] = (bf16_t)0.0f; }
    }
  };
  auto loadB = [&](int k0, v8bf& y0, v8bf& y1) {
    const bf16_t* p = Bm + (size_t)(k0 + kB) * N + n0 + nB;
    y0 = *(const v8bf*)(p);
    y1 = *(const v8bf*)(p + 8);
  };

  v8bf a0, a1, b0, b1;
  loadA(0, a0, a1);
  loadB(0, b0, b1);

  for (int k0 = 0; k0 < K; k0 += BK) {
    __syncthreads();  // previous iteration's LDS reads complete
    // ---- commit staged registers to LDS ----
    *(v8bf*)(&As[rowA * LDP + colA])     = a0;
    *(v8bf*)(&As[rowA * LDP + colA + 8]) = a1;
#pragma unroll
    for (int e = 0; e < 8; ++e) {
      Bs[(nB + e) * LDP + kB]     = b0[e];
      Bs[(nB + 8 + e) * LDP + kB] = b1[e];
    }
    __syncthreads();
    // ---- issue next tile's global loads; they complete under the WMMAs ----
    if (k0 + BK < K) {
      loadA(k0 + BK, a0, a1);
      loadB(k0 + BK, b0, b1);
    }

    // ---- A fragment (one per wave per K-step) ----
    v16bf afr;
    {
      v8bf lo = *(const v8bf*)(&As[mrow * LDP + kbA]);
      v8bf hi = *(const v8bf*)(&As[mrow * LDP + 16 + kbA]);
#pragma unroll
      for (int e = 0; e < 8; ++e) { afr[e] = lo[e]; afr[8 + e] = hi[e]; }
    }
    // ---- load all 8 B fragments, then the WMMA burst ----
    v16bf bfr[8];
#pragma unroll
    for (int ns = 0; ns < 8; ++ns) {
      int coln = ns * 16 + lane15;
      v8bf lo = *(const v8bf*)(&Bs[coln * LDP + kbB]);
      v8bf hi = *(const v8bf*)(&Bs[coln * LDP + kbB + 16]);
#pragma unroll
      for (int e = 0; e < 8; ++e) { bfr[ns][e] = lo[e]; bfr[ns][8 + e] = hi[e]; }
    }
#pragma unroll
    for (int ns = 0; ns < 8; ++ns) {
      acc[ns] = __builtin_amdgcn_wmma_f32_16x16x32_bf16(
          false, afr, false, bfr[ns], (short)0, acc[ns], false, false);
    }
  }

  // ---- epilogue: C layout per ISA (vgpr r: M=r / M=8+r, N=lane&15) ----
#pragma unroll
  for (int ns = 0; ns < 8; ++ns) {
#pragma unroll
    for (int r = 0; r < 8; ++r) {
      int grow = m0 + wv * 16 + ((lane < 16) ? r : 8 + r);
      int gcol = n0 + ns * 16 + lane15;
      if (!GUARD || grow < M) {
        float v = acc[ns][r];
        if (bias) v += bias[gcol];
        if (ACT == ACT_RELU)      v = fmaxf(v, 0.0f);
        else if (ACT == ACT_GELU) v = 0.5f * v * (1.0f + erff(v * 0.70710678118f));
        else if (ACT == ACT_SIG)  v = 1.0f / (1.0f + __expf(-v));
        if (Cf) Cf[(size_t)grow * N + gcol] = v;
        if (Cb) Cb[(size_t)grow * N + gcol] = (bf16_t)v;
      }
    }
  }
}

// row sums of relu'd A_dyn.  grid = N
__global__ __launch_bounds__(256) void rowsum_kernel(
    const float* __restrict__ Adyn, float* __restrict__ rs) {
  __shared__ float red[256];
  int i = blockIdx.x, tid = threadIdx.x;
  float s = 0.0f;
  for (int j = tid; j < NN; j += 256) s += Adyn[(size_t)i * NN + j];
  red[tid] = s; __syncthreads();
  for (int off = 128; off > 0; off >>= 1) {
    if (tid < off) red[tid] += red[tid + off];
    __syncthreads();
  }
  if (tid == 0) rs[i] = red[0];
}

// g[n] = mean_b gamma[b][n].  grid = N/256
__global__ void gmean_kernel(const float* __restrict__ gamma,
                             float* __restrict__ g) {
  int n = blockIdx.x * blockDim.x + threadIdx.x;
  float s = 0.0f;
  for (int b = 0; b < BB; ++b) s += gamma[(size_t)b * NN + n];
  g[n] = s * (1.0f / BB);
}

// A_mix^T (bf16) = [alpha*Ap + (1-alpha)*0.5*(An_ij + An_ji)] * g[i], stored
// transposed for the z_gcn GEMM. blockIdx.y = j, threads over i (coalesced
// writes; strided reads stay in the 192MB L2 — A_dyn is only 4MB).
__global__ void abuild_kernel(
    const float* __restrict__ Ap, const float* __restrict__ Adyn,
    const float* __restrict__ rs, const float* __restrict__ g,
    const float* __restrict__ alpha_raw, bf16_t* __restrict__ AmixT) {
  int i = blockIdx.x * blockDim.x + threadIdx.x;
  int j = blockIdx.y;
  float alpha = 1.0f / (1.0f + __expf(-alpha_raw[0]));
  float aij = Adyn[(size_t)i * NN + j] / (rs[i] + 1e-6f);
  float aji = Adyn[(size_t)j * NN + i] / (rs[j] + 1e-6f);
  float ab = alpha * Ap[(size_t)i * NN + j] + (1.0f - alpha) * 0.5f * (aij + aji);
  AmixT[(size_t)j * NN + i] = (bf16_t)(ab * g[i]);
}

// y = h*(1+gw) + msg; LayerNorm over H.  grid = B*T, block = 256 (4/thread)
__global__ __launch_bounds__(256) void final_ln_kernel(
    const float* __restrict__ h, const float* __restrict__ msg,
    const float* __restrict__ lnw, const float* __restrict__ lnb,
    float* __restrict__ out, float gw) {
  __shared__ float r1[256], r2[256];
  size_t base = (size_t)blockIdx.x * HH;
  int tid = threadIdx.x;
  float y[4]; float s = 0.0f, ss = 0.0f;
#pragma unroll
  for (int i = 0; i < 4; ++i) {
    int hi = tid + i * 256;
    float v = h[base + hi] * (1.0f + gw) + msg[base + hi];
    y[i] = v; s += v; ss += v * v;
  }
  r1[tid] = s; r2[tid] = ss; __syncthreads();
  for (int off = 128; off > 0; off >>= 1) {
    if (tid < off) { r1[tid] += r1[tid + off]; r2[tid] += r2[tid + off]; }
    __syncthreads();
  }
  float mu   = r1[0] * (1.0f / HH);
  float var  = r2[0] * (1.0f / HH) - mu * mu;
  float rstd = rsqrtf(var + 1e-5f);
#pragma unroll
  for (int i = 0; i < 4; ++i) {
    int hi = tid + i * 256;
    out[base + hi] = (y[i] - mu) * rstd * lnw[hi] + lnb[hi];
  }
}

// ---------------------------------------------------------------------------
static void launch_gemm(int act, const bf16_t* A, const bf16_t* B,
                        const float* bias, float* Cf, bf16_t* Cb,
                        int M, int N, int K, hipStream_t s) {
  dim3 grid(N / 128, (M + 127) / 128);
  if (M % 128 == 0) {
    switch (act) {
      case ACT_RELU: gemm_bf16_kernel<ACT_RELU,false><<<grid,256,0,s>>>(A,B,bias,Cf,Cb,M,N,K); break;
      case ACT_GELU: gemm_bf16_kernel<ACT_GELU,false><<<grid,256,0,s>>>(A,B,bias,Cf,Cb,M,N,K); break;
      case ACT_SIG:  gemm_bf16_kernel<ACT_SIG ,false><<<grid,256,0,s>>>(A,B,bias,Cf,Cb,M,N,K); break;
      default:       gemm_bf16_kernel<ACT_NONE,false><<<grid,256,0,s>>>(A,B,bias,Cf,Cb,M,N,K); break;
    }
  } else {
    switch (act) {
      case ACT_RELU: gemm_bf16_kernel<ACT_RELU,true><<<grid,256,0,s>>>(A,B,bias,Cf,Cb,M,N,K); break;
      case ACT_GELU: gemm_bf16_kernel<ACT_GELU,true><<<grid,256,0,s>>>(A,B,bias,Cf,Cb,M,N,K); break;
      case ACT_SIG:  gemm_bf16_kernel<ACT_SIG ,true><<<grid,256,0,s>>>(A,B,bias,Cf,Cb,M,N,K); break;
      default:       gemm_bf16_kernel<ACT_NONE,true><<<grid,256,0,s>>>(A,B,bias,Cf,Cb,M,N,K); break;
    }
  }
}

extern "C" void kernel_launch(void* const* d_in, const int* in_sizes, int n_in,
                              void* d_out, int out_size, void* d_ws, size_t ws_size,
                              hipStream_t stream) {
  (void)in_sizes; (void)n_in; (void)out_size; (void)ws_size;
  const float* h_time    = (const float*)d_in[0];
  const float* A_pearson = (const float*)d_in[1];
  const float* E_dyn     = (const float*)d_in[2];
  const float* alpha_raw = (const float*)d_in[3];
  /* d_in[4] = tau_raw: unused — softmax(axis=0).mean(axis=0) == 1/N exactly */
  const float* gate_W1 = (const float*)d_in[5];
  const float* gate_b1 = (const float*)d_in[6];
  const float* gate_W2 = (const float*)d_in[7];
  const float* gate_b2 = (const float*)d_in[8];
  const float* h2n_W   = (const float*)d_in[9];
  const float* h2n_b   = (const float*)d_in[10];
  const float* n2h_W   = (const float*)d_in[11];
  const float* n2h_b   = (const float*)d_in[12];
  const float* ln_w    = (const float*)d_in[13];
  const float* ln_b    = (const float*)d_in[14];
  float* out = (float*)d_out;

  // ---- workspace carve-out (~190 MB) ----
  char* ws = (char*)d_ws; size_t off = 0;
  auto carve = [&](size_t bytes) -> char* {
    char* p = ws + off; off = (off + bytes + 255) & ~(size_t)255; return p;
  };
  const size_t BT = (size_t)BB * TT;                       // 16384
  bf16_t* h_b     = (bf16_t*)carve(BT * HH * 2);           // bf16 h_time
  bf16_t* zn_b    = (bf16_t*)carve(BT * NN * 2);           // z_nodes (bf16)
  bf16_t* zg_b    = (bf16_t*)carve(BT * NN * 2);           // z_gcn (bf16)
  float*  msg     = (float*) carve(BT * HH * 4);           // msg_h (f32)
  float*  hgate_f = (float*) carve((size_t)BB * HH * 4);
  bf16_t* hgate_b = (bf16_t*)carve((size_t)BB * HH * 2);
  float*  hmean   = (float*) carve(HH * 4);
  bf16_t* Ecur    = (bf16_t*)carve((size_t)NN * HH * 2);
  bf16_t* EcurT   = (bf16_t*)carve((size_t)NN * HH * 2);
  float*  Adyn    = (float*) carve((size_t)NN * NN * 4);
  float*  rs      = (float*) carve(NN * 4);
  bf16_t* W1b     = (bf16_t*)carve((size_t)HH * (HH/2) * 2);
  bf16_t* W2b     = (bf16_t*)carve((size_t)(HH/2) * NN * 2);
  bf16_t* h2nWb   = (bf16_t*)carve((size_t)HH * NN * 2);
  bf16_t* n2hWb   = (bf16_t*)carve((size_t)NN * HH * 2);
  bf16_t* hidb    = (bf16_t*)carve((size_t)BB * (HH/2) * 2);
  float*  gamma   = (float*) carve((size_t)BB * NN * 4);
  float*  g       = (float*) carve(NN * 4);
  bf16_t* AmixT   = (bf16_t*)carve((size_t)NN * NN * 2);

  // 1) reductions + bf16 conversion of h_time (single pass over 64MB)
  reduce_bt_kernel<<<dim3(HH/256, BB), 256, 0, stream>>>(h_time, hgate_f, hgate_b, h_b);
  reduce_b_kernel<<<HH/256, 256, 0, stream>>>(hgate_f, hmean);

  // 2) weight conversions to bf16
  cvt_bf16_kernel<<<(HH*(HH/2)+255)/256, 256, 0, stream>>>(gate_W1, W1b, HH*(HH/2));
  cvt_bf16_kernel<<<((HH/2)*NN+255)/256, 256, 0, stream>>>(gate_W2, W2b, (HH/2)*NN);
  cvt_bf16_kernel<<<(HH*NN+255)/256,     256, 0, stream>>>(h2n_W,  h2nWb, HH*NN);
  cvt_bf16_kernel<<<(NN*HH+255)/256,     256, 0, stream>>>(n2h_W,  n2hWb, NN*HH);

  // 3) E_cur (normalized) + transpose, then A_dyn = relu(E_cur @ E_cur^T)
  ecur_kernel<<<NN, 256, 0, stream>>>(E_dyn, hmean, Ecur, EcurT);
  launch_gemm(ACT_RELU, Ecur, EcurT, nullptr, Adyn, nullptr, NN, NN, HH, stream);
  rowsum_kernel<<<NN, 256, 0, stream>>>(Adyn, rs);

  // 4) gate MLP -> gamma -> g
  launch_gemm(ACT_GELU, hgate_b, W1b, gate_b1, nullptr, hidb, BB, HH/2, HH, stream);
  launch_gemm(ACT_SIG,  hidb,    W2b, gate_b2, gamma,  nullptr, BB, NN, HH/2, stream);
  gmean_kernel<<<NN/256, 256, 0, stream>>>(gamma, g);

  // 5) A_mix^T (normalize + symmetrize + pearson blend + row-scale, bf16)
  abuild_kernel<<<dim3(NN/256, NN), 256, 0, stream>>>(A_pearson, Adyn, rs, g,
                                                      alpha_raw, AmixT);

  // 6) heavy chain: 3 x (16384 x 1024 x 1024) bf16 WMMA GEMMs
  launch_gemm(ACT_NONE, h_b,  h2nWb, h2n_b,   nullptr, zn_b, (int)BT, NN, HH, stream);
  launch_gemm(ACT_NONE, zn_b, AmixT, nullptr, nullptr, zg_b, (int)BT, NN, NN, stream);
  launch_gemm(ACT_NONE, zg_b, n2hWb, n2h_b,   msg,     nullptr, (int)BT, HH, NN, stream);

  // 7) residual + constant global gate (tanh(1/N)) + LayerNorm
  float gw = tanhf(1.0f / (float)NN);
  final_ln_kernel<<<(int)BT, 256, 0, stream>>>(h_time, msg, ln_w, ln_b, out, gw);
}
